// ReDialRecModel_10093173145846
// MI455X (gfx1250) — matrix operand order repaired and checked by reading, not compiled, not run on hardware
//
#include <hip/hip_runtime.h>
#include <hip/hip_bf16.h>

typedef __attribute__((ext_vector_type(16))) __bf16 v16bf;
typedef __attribute__((ext_vector_type(8)))  float  v8f;

#define EMB 128
#define NB  4096
#define NE  30000

// ---------- bf16 helpers (bit-exact RNE split, no reliance on __bf16 arithmetic) ----------
static __device__ __forceinline__ __bf16 f32_to_bf16_rne(float f) {
  unsigned u = __builtin_bit_cast(unsigned, f);
  unsigned r = u + 0x7FFFu + ((u >> 16) & 1u);   // round-to-nearest-even
  unsigned short h = (unsigned short)(r >> 16);
  return __builtin_bit_cast(__bf16, h);
}
static __device__ __forceinline__ float bf16_to_f32(__bf16 b) {
  unsigned short h = __builtin_bit_cast(unsigned short, b);
  unsigned u = ((unsigned)h) << 16;
  return __builtin_bit_cast(float, u);
}

static __device__ __forceinline__ int lower_bound_i32(const int* __restrict__ a, int n, int v) {
  int lo = 0, hi = n;
  while (lo < hi) { int mid = (lo + hi) >> 1; if (a[mid] < v) lo = mid + 1; else hi = mid; }
  return lo;
}

// ---------- Kernel 1: segment-mean pool both tables, emit user as A-fragments (hi/lo bf16) ----
// A 16x32 bf16 WMMA layout (ISA 7.12.2): lane l<16 -> row M=l, vec elems 0..7 = K 0..7,
// elems 8..15 = K 16..23 ; lane l>=16 -> row M=l-16, K 8..15 and 24..31.
__global__ void pool_user_frag(const float* __restrict__ ent_emb, const float* __restrict__ itm_emb,
                               const int* __restrict__ eidx, const int* __restrict__ eseg,
                               const int* __restrict__ iidx, const int* __restrict__ iseg,
                               int T, __bf16* __restrict__ ApHi, __bf16* __restrict__ ApLo) {
  const int b = blockIdx.x;        // batch row (segment id)
  const int d = threadIdx.x;       // embedding dim 0..127
  const int elo = lower_bound_i32(eseg, T, b), ehi = lower_bound_i32(eseg, T, b + 1);
  const int ilo = lower_bound_i32(iseg, T, b), ihi = lower_bound_i32(iseg, T, b + 1);
  float se = 0.f;
  for (int t = elo; t < ehi; ++t) se += ent_emb[(size_t)eidx[t] * EMB + d];
  float si = 0.f;
  for (int t = ilo; t < ihi; ++t) si += itm_emb[(size_t)iidx[t] * EMB + d];
  const int ce = ehi - elo, ci = ihi - ilo;
  const float u = 0.5f * (se / (float)(ce > 0 ? ce : 1) + si / (float)(ci > 0 ? ci : 1));
  // scatter into A-fragment layout
  const int r = b & 15, tm = b >> 4;
  const int c = d >> 5, dk = d & 31;
  const int l = r + (((dk >> 3) & 1) << 4);
  const int i = (dk & 7) + (dk >= 16 ? 8 : 0);
  const int idx = ((tm * 4 + c) * 32 + l) * 16 + i;
  const __bf16 hi = f32_to_bf16_rne(u);
  ApHi[idx] = hi;
  ApLo[idx] = f32_to_bf16_rne(u - bf16_to_f32(hi));
}

// ---------- Kernel 2: convert W_dec into B-fragments (hi/lo bf16) ----------
// B 32x16 bf16 layout: lane n<16 -> col N=n, vec elems = K 0..7,16..23 ; lane n>=16 -> K 8..15,24..31.
__global__ void conv_w_frag(const float* __restrict__ W,
                            __bf16* __restrict__ BpHi, __bf16* __restrict__ BpLo) {
  const int gid = blockIdx.x * blockDim.x + threadIdx.x;
  if (gid >= NE * EMB) return;
  const int n = gid >> 7, k = gid & 127;      // W is [NE][128] row-major
  const float x = W[gid];
  const int tn = n >> 4, r = n & 15, c = k >> 5, dk = k & 31;
  const int l = r + (((dk >> 3) & 1) << 4);
  const int i = (dk & 7) + (dk >= 16 ? 8 : 0);
  const int idx = ((tn * 4 + c) * 32 + l) * 16 + i;
  const __bf16 hi = f32_to_bf16_rne(x);
  BpHi[idx] = hi;
  BpLo[idx] = f32_to_bf16_rne(x - bf16_to_f32(hi));
}

// ---------- Kernel 3: GEMM logits = user @ W^T + bias, bf16x3 split WMMA ----------
// Each wave: 2 (M) x 3 (N) tiles of 16x16, K=128 as 4 chunks of 32, 3 WMMA terms each
// -> 72 v_wmma_f32_16x16x32_bf16 per wave. 8 waves/block.
__global__ __launch_bounds__(256) void gemm_bf16x3(
    const v16bf* __restrict__ ApHi, const v16bf* __restrict__ ApLo,
    const v16bf* __restrict__ BpHi, const v16bf* __restrict__ BpLo,
    const float* __restrict__ bias, float* __restrict__ out) {
  const int lane = threadIdx.x & 31;
  const int wave = threadIdx.x >> 5;
  const int mrow = blockIdx.y * 8 + wave;   // 0..127 wave rows (32 M each)
  const int mt0  = mrow * 2;                // first of 2 M tiles
  const int nt0  = blockIdx.x * 3;          // first of 3 N tiles (48 N per block-col)

  v8f acc[2][3];
#pragma unroll
  for (int mi = 0; mi < 2; ++mi)
#pragma unroll
    for (int ni = 0; ni < 3; ++ni) acc[mi][ni] = (v8f)0.0f;

#pragma unroll
  for (int c = 0; c < 4; ++c) {
    v16bf aH[2], aL[2], bH[3], bL[3];
#pragma unroll
    for (int mi = 0; mi < 2; ++mi) {
      const int o = ((mt0 + mi) * 4 + c) * 32 + lane;  // v16bf units (32B/lane, coalesced)
      aH[mi] = ApHi[o];
      aL[mi] = ApLo[o];
    }
#pragma unroll
    for (int ni = 0; ni < 3; ++ni) {
      const int o = ((nt0 + ni) * 4 + c) * 32 + lane;
      bH[ni] = BpHi[o];
      bL[ni] = BpLo[o];
    }
#pragma unroll
    for (int mi = 0; mi < 2; ++mi)
#pragma unroll
      for (int ni = 0; ni < 3; ++ni) {
        acc[mi][ni] = __builtin_amdgcn_wmma_f32_16x16x32_bf16(
            false, aH[mi], false, bH[ni], (short)0, acc[mi][ni], false, false);
        acc[mi][ni] = __builtin_amdgcn_wmma_f32_16x16x32_bf16(
            false, aH[mi], false, bL[ni], (short)0, acc[mi][ni], false, false);
        acc[mi][ni] = __builtin_amdgcn_wmma_f32_16x16x32_bf16(
            false, aL[mi], false, bH[ni], (short)0, acc[mi][ni], false, false);
      }
  }

  // C/D layout: VGPR e, lane<16 -> (M = e, N = lane); lane>=16 -> (M = e+8, N = lane-16)
  const int colw = lane & 15;
  const int rsel = (lane >> 4) << 3;
#pragma unroll
  for (int ni = 0; ni < 3; ++ni) {
    const int col = (nt0 + ni) * 16 + colw;
    const float bv = bias[col];
#pragma unroll
    for (int mi = 0; mi < 2; ++mi) {
      const int row0 = (mt0 + mi) * 16 + rsel;
#pragma unroll
      for (int e = 0; e < 8; ++e)
        out[(size_t)(row0 + e) * NE + col] = acc[mi][ni][e] + bv;
    }
  }
}

// ---------- Kernel 4: per-row online logsumexp + per-row NLL, pass labels through ----------
__global__ void row_logsumexp_loss(const float* __restrict__ logits, const int* __restrict__ labels,
                                   float* __restrict__ rowloss, float* __restrict__ out_labels) {
  __shared__ float sm[256], ss[256];
  const int b = blockIdx.x, tid = threadIdx.x;
  const float* rowp = logits + (size_t)b * NE;
  float m = -__builtin_inff(), s = 0.f;
  for (int j = tid; j < NE; j += 256) {
    const float x = rowp[j];
    const float nm = fmaxf(m, x);
    s = s * expf(m - nm) + expf(x - nm);
    m = nm;
  }
  sm[tid] = m; ss[tid] = s;
  __syncthreads();
  for (int off = 128; off > 0; off >>= 1) {
    if (tid < off) {
      const float m1 = sm[tid], s1 = ss[tid];
      const float m2 = sm[tid + off], s2 = ss[tid + off];
      const float nm = fmaxf(m1, m2);
      sm[tid] = nm;
      ss[tid] = s1 * expf(m1 - nm) + s2 * expf(m2 - nm);
    }
    __syncthreads();
  }
  if (tid == 0) {
    const float lse = sm[0] + logf(ss[0]);
    const int lab = labels[b];
    rowloss[b] = lse - rowp[lab];        // -log_softmax at label
    out_labels[b] = (float)lab;
  }
}

// ---------- Kernel 5: mean over rows ----------
__global__ void mean_loss(const float* __restrict__ rowloss, float* __restrict__ out_loss) {
  __shared__ float sh[256];
  const int tid = threadIdx.x;
  float s = 0.f;
  for (int b = tid; b < NB; b += 256) s += rowloss[b];
  sh[tid] = s;
  __syncthreads();
  for (int off = 128; off > 0; off >>= 1) {
    if (tid < off) sh[tid] += sh[tid + off];
    __syncthreads();
  }
  if (tid == 0) *out_loss = sh[0] / (float)NB;
}

extern "C" void kernel_launch(void* const* d_in, const int* in_sizes, int n_in,
                              void* d_out, int out_size, void* d_ws, size_t ws_size,
                              hipStream_t stream) {
  const float* ent_emb = (const float*)d_in[0];
  const float* itm_emb = (const float*)d_in[1];
  const float* W       = (const float*)d_in[2];
  const float* bias    = (const float*)d_in[3];
  const int*   eidx    = (const int*)d_in[4];
  const int*   eseg    = (const int*)d_in[5];
  const int*   iidx    = (const int*)d_in[6];
  const int*   iseg    = (const int*)d_in[7];
  const int*   labels  = (const int*)d_in[8];
  const int T = in_sizes[4];

  // workspace layout (bytes): A frags 2x1MB, B frags 2x7.68MB, rowloss 16KB  (~17.5 MB)
  char* ws = (char*)d_ws;
  __bf16* ApHi = (__bf16*)(ws);
  __bf16* ApLo = (__bf16*)(ws + 1048576);
  __bf16* BpHi = (__bf16*)(ws + 2097152);
  __bf16* BpLo = (__bf16*)(ws + 2097152 + 7680000);
  float*  rowloss = (float*)(ws + 2097152 + 2 * 7680000);

  float* logits     = (float*)d_out;
  float* out_labels = logits + (size_t)NB * NE;
  float* out_loss   = out_labels + NB;

  pool_user_frag<<<NB, EMB, 0, stream>>>(ent_emb, itm_emb, eidx, eseg, iidx, iseg, T, ApHi, ApLo);
  conv_w_frag<<<(NE * EMB + 255) / 256, 256, 0, stream>>>(W, BpHi, BpLo);
  dim3 g(NE / 48, NB / 256);  // 625 x 16 blocks, 256 threads (8 waves)
  gemm_bf16x3<<<g, 256, 0, stream>>>((const v16bf*)ApHi, (const v16bf*)ApLo,
                                     (const v16bf*)BpHi, (const v16bf*)BpLo, bias, logits);
  row_logsumexp_loss<<<NB, 256, 0, stream>>>(logits, labels, rowloss, out_labels);
  mean_loss<<<1, 256, 0, stream>>>(rowloss, out_loss);
}
